// OptimalTransportTrafficElements_41755672052332
// MI455X (gfx1250) — matrix operand order repaired and checked by reading, not compiled, not run on hardware
//
#include <hip/hip_runtime.h>
#include <hip/hip_bf16.h>

// ---------------------------------------------------------------------------
// Problem constants (from reference): B=4, N_LC=N_TE=2048, D=256, ITERS=100
// ---------------------------------------------------------------------------
#define BATCH 4
#define NQ    2048
#define DIM   256
#define NAUG  2049              // 2048 + dustbin
#define LDZ   2049              // Z stored directly in d_out, stride = row len
#define LDUV  2064              // padded stride for u/v vectors
#define SINK_ITERS 100
#define SINK_BLOCKS 260         // = 4 batches * ceil(2049/32) col strips

#define NORM_CONST  (-8.3177661667193433f)   // -log(4096)
#define LOG_N_CONST ( 7.6246189861593985f)   //  log(2048)

typedef __attribute__((ext_vector_type(16))) __bf16 v16bf;
typedef __attribute__((ext_vector_type(8)))  __bf16 v8bf;
typedef __attribute__((ext_vector_type(8)))  float  v8f;

// ---------------------------------------------------------------------------
// CDNA5 async global->LDS copy (ASYNCcnt-tracked; bypasses VGPRs).
// LDS byte address = low 32 bits of the generic pointer (flat LDS aperture).
// ---------------------------------------------------------------------------
__device__ __forceinline__ void async_copy_b128(const void* gptr, void* lptr)
{
    const unsigned long long ga = (unsigned long long)gptr;
    const unsigned int       la = (unsigned int)(unsigned long long)lptr;
    asm volatile("global_load_async_to_lds_b128 %0, %1, off"
                 :: "v"(la), "v"(ga) : "memory");
}
__device__ __forceinline__ void async_wait0()
{
    asm volatile("s_wait_asynccnt 0" ::: "memory");
}

// ---------------------------------------------------------------------------
// WMMA GEMM:  C[n,k] = act( scale * sum_d A[n,d]*Bt[k,d] + bias[k] )
// A: [M,K] row-major (f32 or bf16), Bt: [N,K] row-major (weights / f2),
// i.e. C = A @ Bt^T.  Block tile 128x128, K-step 32, 8 waves (2x4) per block,
// each wave does 4x2 tiles of v_wmma_f32_16x16x32_bf16.
// bf16 operands stage through LDS via global_load_async_to_lds_b128;
// f32 operands go through VGPRs for the f32->bf16 convert.
// ---------------------------------------------------------------------------
#define BM 128
#define BN 128
#define BK 32
#define LDS_STR 40   // bf16 elems per LDS row (80B) -> conflict-free b128 reads

template <bool A_F32, bool B_F32, bool OUT_BF16>
__global__ __launch_bounds__(256)
void gemm_wmma(const void* __restrict__ Ap, const void* __restrict__ Bp,
               void* __restrict__ Cp, const float* __restrict__ bias,
               float scale, int relu, int K, int lda, int ldb, int ldc,
               long sA, long sB, long sC)
{
    __shared__ __align__(16) __bf16 As[BM][LDS_STR];
    __shared__ __align__(16) __bf16 Bs[BN][LDS_STR];

    const int t    = threadIdx.x;
    const int lane = t & 31;
    const int wave = t >> 5;
    const int wM   = (wave & 1) * 64;       // wave grid: 2 (M) x 4 (N)
    const int wN   = (wave >> 1) * 32;
    const int rowBase = blockIdx.y * BM;
    const int colBase = blockIdx.x * BN;
    const int z       = blockIdx.z;

    const float*  Af = (const float*)Ap  + (size_t)z * sA;
    const __bf16* Ah = (const __bf16*)Ap + (size_t)z * sA;
    const float*  Bf = (const float*)Bp  + (size_t)z * sB;
    const __bf16* Bh = (const __bf16*)Bp + (size_t)z * sB;

    // staging assignment: 2 threads per row, 16 elements each
    const int sr = t >> 1;
    const int sc = (t & 1) * 16;

    v8f acc[4][2];
#pragma unroll
    for (int mt = 0; mt < 4; ++mt)
#pragma unroll
        for (int nt = 0; nt < 2; ++nt) acc[mt][nt] = (v8f)0.0f;

    for (int k0 = 0; k0 < K; k0 += BK) {
        __syncthreads();
        // ---- stage A tile ----
        if constexpr (A_F32) {
            const float* p = Af + (size_t)(rowBase + sr) * lda + k0 + sc;
            if (k0 + BK < K) __builtin_prefetch(p + BK, 0, 0);  // global_prefetch_b8
            __align__(16) __bf16 tmp[16];
#pragma unroll
            for (int q = 0; q < 4; ++q) {
                float4 f = *(const float4*)(p + q * 4);
                tmp[q*4+0] = (__bf16)f.x; tmp[q*4+1] = (__bf16)f.y;
                tmp[q*4+2] = (__bf16)f.z; tmp[q*4+3] = (__bf16)f.w;
            }
            *(v8bf*)&As[sr][sc]     = *(const v8bf*)&tmp[0];
            *(v8bf*)&As[sr][sc + 8] = *(const v8bf*)&tmp[8];
        } else {
            const __bf16* p = Ah + (size_t)(rowBase + sr) * lda + k0 + sc;
            async_copy_b128(p,     &As[sr][sc]);
            async_copy_b128(p + 8, &As[sr][sc + 8]);
        }
        // ---- stage B tile (rows of Bs are output columns) ----
        if constexpr (B_F32) {
            const float* p = Bf + (size_t)(colBase + sr) * ldb + k0 + sc;
            if (k0 + BK < K) __builtin_prefetch(p + BK, 0, 0);
            __align__(16) __bf16 tmp[16];
#pragma unroll
            for (int q = 0; q < 4; ++q) {
                float4 f = *(const float4*)(p + q * 4);
                tmp[q*4+0] = (__bf16)f.x; tmp[q*4+1] = (__bf16)f.y;
                tmp[q*4+2] = (__bf16)f.z; tmp[q*4+3] = (__bf16)f.w;
            }
            *(v8bf*)&Bs[sr][sc]     = *(const v8bf*)&tmp[0];
            *(v8bf*)&Bs[sr][sc + 8] = *(const v8bf*)&tmp[8];
        } else {
            const __bf16* p = Bh + (size_t)(colBase + sr) * ldb + k0 + sc;
            async_copy_b128(p,     &Bs[sr][sc]);
            async_copy_b128(p + 8, &Bs[sr][sc + 8]);
        }
        if constexpr (!A_F32 || !B_F32) async_wait0();
        __syncthreads();

        // ---- load fragments per the CDNA5 16-bit WMMA lane layouts ----
        // A 16x32: lane<16 -> K 0..7 / 16..23, lane>=16 -> K 8..15 / 24..31
        // B 32x16: lane<16 -> K 0..7 / 16..23, lane>=16 -> K 8..15 / 24..31
        const int am  = wM + (lane & 15);
        const int alo = (lane & 16) ? 8 : 0;
        const int bn  = wN + (lane & 15);
        const int bko = (lane & 16) ? 8 : 0;

        union Frag { v16bf v; v8bf h[2]; };
        Frag a[4], b[2];
#pragma unroll
        for (int mt = 0; mt < 4; ++mt) {
            a[mt].h[0] = *(const v8bf*)&As[am + mt * 16][alo];
            a[mt].h[1] = *(const v8bf*)&As[am + mt * 16][16 + alo];
        }
#pragma unroll
        for (int nt = 0; nt < 2; ++nt) {
            b[nt].h[0] = *(const v8bf*)&Bs[bn + nt * 16][bko];
            b[nt].h[1] = *(const v8bf*)&Bs[bn + nt * 16][bko + 16];
        }
#pragma unroll
        for (int mt = 0; mt < 4; ++mt)
#pragma unroll
            for (int nt = 0; nt < 2; ++nt)
                acc[mt][nt] = __builtin_amdgcn_wmma_f32_16x16x32_bf16(
                    false, a[mt].v, false, b[nt].v,
                    (short)0, acc[mt][nt], false, false);
    }

    // ---- epilogue: C/D layout VGPR r = row r (+8 for upper half-wave) ----
    const int rsel = (lane & 16) ? 8 : 0;
#pragma unroll
    for (int mt = 0; mt < 4; ++mt) {
#pragma unroll
        for (int nt = 0; nt < 2; ++nt) {
            const int col = colBase + wN + nt * 16 + (lane & 15);
#pragma unroll
            for (int r = 0; r < 8; ++r) {
                const int row = rowBase + wM + mt * 16 + rsel + r;
                float val = acc[mt][nt][r] * scale;
                if constexpr (OUT_BF16) {
                    if (bias) val += bias[col];
                    if (relu) val = fmaxf(val, 0.0f);
                    ((__bf16*)Cp)[(size_t)z * sC + (size_t)row * ldc + col] = (__bf16)val;
                } else {
                    ((float*)Cp)[(size_t)z * sC + (size_t)row * ldc + col] = val;
                }
            }
        }
    }
}

// ---------------------------------------------------------------------------
// Init: zero u/v, fill dustbin row/col of Z with alpha, reset barrier counter
// ---------------------------------------------------------------------------
__global__ void init_kernel(float* __restrict__ Z, float* __restrict__ u,
                            float* __restrict__ v, int* __restrict__ bar,
                            const float* __restrict__ alphap)
{
    const float alpha = *alphap;
    const int idx = blockIdx.x * blockDim.x + threadIdx.x;
    if (idx < BATCH * LDUV) { u[idx] = 0.0f; v[idx] = 0.0f; }
    if (idx < BATCH * NAUG) {
        const int b = idx / NAUG, r = idx - b * NAUG;
        float* Zb = Z + (size_t)b * NAUG * LDZ;
        Zb[(size_t)r * LDZ + NQ] = alpha;          // last column (incl. corner)
        Zb[(size_t)NQ * LDZ + r] = alpha;          // last row
    }
    if (idx == 0) bar[0] = 0;                      // gen (bar[1]) is relative
}

// ---------------------------------------------------------------------------
// Device-wide sense-reversing barrier (persistent cooperative kernel)
// ---------------------------------------------------------------------------
__device__ __forceinline__ void grid_barrier(int* cnt, int* gen, int nblk)
{
    __syncthreads();
    if (threadIdx.x == 0) {
        __threadfence();
        const int g = __hip_atomic_load(gen, __ATOMIC_RELAXED, __HIP_MEMORY_SCOPE_AGENT);
        const int prev = __hip_atomic_fetch_add(cnt, 1, __ATOMIC_ACQ_REL, __HIP_MEMORY_SCOPE_AGENT);
        if (prev == nblk - 1) {
            __hip_atomic_store(cnt, 0, __ATOMIC_RELAXED, __HIP_MEMORY_SCOPE_AGENT);
            __hip_atomic_store(gen, g + 1, __ATOMIC_RELEASE, __HIP_MEMORY_SCOPE_AGENT);
        } else {
            while (__hip_atomic_load(gen, __ATOMIC_ACQUIRE, __HIP_MEMORY_SCOPE_AGENT) == g)
                __builtin_amdgcn_s_sleep(2);
        }
        __threadfence();
    }
    __syncthreads();
}

// ---------------------------------------------------------------------------
// Persistent Sinkhorn: 100 iterations, 2 device barriers each.
// Two-pass logsumexp (max sweep, then sum-exp sweep): 1 v_exp_f32 per
// element instead of 2, and the loop-carried chain is just v_max / v_add —
// Z (67MB) is L2-resident (192MB L2) so the second sweep is cheap.
// u-pass: one row per wave (lanes stride cols, 128B coalesced).
// v-pass: 32-col strip per block, 8 row-groups/col (128B coalesced rows),
//         LDS combine of the 8 partials.
// ---------------------------------------------------------------------------
__global__ __launch_bounds__(256)
void sinkhorn_kernel(const float* __restrict__ Z, float* __restrict__ u,
                     float* __restrict__ v, int* __restrict__ bar)
{
    __shared__ float smM[8][32];
    __shared__ float smS[8][32];
    const int t = threadIdx.x, lane = t & 31, wave = t >> 5;
    int* cnt = bar;
    int* gen = bar + 1;
    const int gw = blockIdx.x * 8 + wave;          // global wave id
    const int NW = SINK_BLOCKS * 8;

    // v-pass strip coordinates (fixed per block)
    const int strip = blockIdx.x;                  // 0..259 = 4 * 65
    const int vb = strip / 65;
    const int vj = (strip - vb * 65) * 32 + lane;  // this thread's column

    for (int it = 0; it < SINK_ITERS; ++it) {
        // ---------------- u = log_mu - LSE_j(Z + v) ----------------
        for (int task = gw; task < BATCH * NAUG; task += NW) {
            const int b = task / NAUG;
            const int i = task - b * NAUG;
            const float* zr = Z + (size_t)b * NAUG * LDZ + (size_t)i * LDZ;
            const float* vr = v + b * LDUV;
            // pass 1: row max
            float M = -1e30f;
            for (int j = lane; j < NAUG; j += 32)
                M = fmaxf(M, zr[j] + vr[j]);
#pragma unroll
            for (int off = 16; off > 0; off >>= 1)
                M = fmaxf(M, __shfl_xor(M, off, 32));
            // pass 2: sum of exp
            float S = 0.0f;
            for (int j = lane; j < NAUG; j += 32)
                S += __expf(zr[j] + vr[j] - M);
#pragma unroll
            for (int off = 16; off > 0; off >>= 1)
                S += __shfl_xor(S, off, 32);
            if (lane == 0) {
                const float lmu = (i < NQ) ? NORM_CONST : (LOG_N_CONST + NORM_CONST);
                u[b * LDUV + i] = lmu - (M + __logf(S));
            }
        }
        grid_barrier(cnt, gen, SINK_BLOCKS);

        // ---------------- v = log_nu - LSE_i(Z + u) ----------------
        {
            const float* zb = Z + (size_t)vb * NAUG * LDZ;
            const float* ur = u + vb * LDUV;
            // pass 1: partial column max per row-group
            float M = -1e30f;
            if (vj < NAUG)
                for (int i = wave; i < NAUG; i += 8)
                    M = fmaxf(M, zb[(size_t)i * LDZ + vj] + ur[i]);
            smM[wave][lane] = M;
            __syncthreads();
            if (wave == 0) {
                float m = smM[0][lane];
#pragma unroll
                for (int g = 1; g < 8; ++g) m = fmaxf(m, smM[g][lane]);
                smM[0][lane] = m;
            }
            __syncthreads();
            const float colM = smM[0][lane];
            // pass 2: partial sum of exp per row-group
            float S = 0.0f;
            if (vj < NAUG)
                for (int i = wave; i < NAUG; i += 8)
                    S += __expf(zb[(size_t)i * LDZ + vj] + ur[i] - colM);
            smS[wave][lane] = S;
            __syncthreads();
            if (wave == 0) {
                float s = smS[0][lane];
#pragma unroll
                for (int g = 1; g < 8; ++g) s += smS[g][lane];
                if (vj < NAUG) {
                    const float lnu = (vj < NQ) ? NORM_CONST : (LOG_N_CONST + NORM_CONST);
                    v[vb * LDUV + vj] = lnu - (colM + __logf(s));
                }
            }
        }
        grid_barrier(cnt, gen, SINK_BLOCKS);
    }
}

// ---------------------------------------------------------------------------
// out = exp(Z + u + v - norm), in place over d_out (Z lives there)
// ---------------------------------------------------------------------------
__global__ __launch_bounds__(256)
void finalize_kernel(float* __restrict__ Z, const float* __restrict__ u,
                     const float* __restrict__ v)
{
    const int j = blockIdx.x * 256 + threadIdx.x;
    const int i = blockIdx.y;
    const int b = blockIdx.z;
    if (j < NAUG) {
        const size_t idx = ((size_t)b * NAUG + i) * LDZ + j;
        Z[idx] = __expf(Z[idx] + u[b * LDUV + i] + v[b * LDUV + j] - NORM_CONST);
    }
}

// ---------------------------------------------------------------------------
extern "C" void kernel_launch(void* const* d_in, const int* in_sizes, int n_in,
                              void* d_out, int out_size, void* d_ws, size_t ws_size,
                              hipStream_t stream)
{
    (void)in_sizes; (void)n_in; (void)out_size; (void)ws_size;
    const float* x_lc  = (const float*)d_in[0];
    const float* x_te  = (const float*)d_in[1];
    const float* W1_lc = (const float*)d_in[2];
    const float* b1_lc = (const float*)d_in[3];
    const float* W2_lc = (const float*)d_in[4];
    const float* b2_lc = (const float*)d_in[5];
    const float* W1_te = (const float*)d_in[6];
    const float* b1_te = (const float*)d_in[7];
    const float* W2_te = (const float*)d_in[8];
    const float* b2_te = (const float*)d_in[9];
    const float* alpha = (const float*)d_in[10];

    float* Z = (float*)d_out;  // augmented score matrix lives in d_out

    // workspace carve-up (~12.7 MB)
    char* p = (char*)d_ws;
    auto carve = [&](size_t bytes) -> void* {
        void* r = (void*)p;
        p += (bytes + 255) & ~(size_t)255;
        return r;
    };
    float*  u   = (float*)carve((size_t)BATCH * LDUV * 4);
    float*  v   = (float*)carve((size_t)BATCH * LDUV * 4);
    __bf16* f1  = (__bf16*)carve((size_t)BATCH * NQ * DIM * 2);
    __bf16* f2  = (__bf16*)carve((size_t)BATCH * NQ * DIM * 2);
    __bf16* h   = (__bf16*)carve((size_t)BATCH * NQ * DIM * 2);
    int*    bar = (int*)carve(256);

    const dim3 blk(256);

    init_kernel<<<(BATCH * LDUV + 255) / 256, blk, 0, stream>>>(Z, u, v, bar, alpha);

    // MLPs: M = B*2048 rows, N = K = 256
    const dim3 gm(DIM / BN, (BATCH * NQ) / BM, 1);
    gemm_wmma<true,  true,  true><<<gm, blk, 0, stream>>>(x_lc, W1_lc, h,  b1_lc, 1.0f, 1, DIM, DIM, DIM, DIM, 0, 0, 0);
    gemm_wmma<false, true,  true><<<gm, blk, 0, stream>>>(h,    W2_lc, f1, b2_lc, 1.0f, 0, DIM, DIM, DIM, DIM, 0, 0, 0);
    gemm_wmma<true,  true,  true><<<gm, blk, 0, stream>>>(x_te, W1_te, h,  b1_te, 1.0f, 1, DIM, DIM, DIM, DIM, 0, 0, 0);
    gemm_wmma<false, true,  true><<<gm, blk, 0, stream>>>(h,    W2_te, f2, b2_te, 1.0f, 0, DIM, DIM, DIM, DIM, 0, 0, 0);

    // scores -> Z interior: per batch, f1[b] @ f2[b]^T / sqrt(256)
    const dim3 gs(NQ / BN, NQ / BM, BATCH);
    gemm_wmma<false, false, false><<<gs, blk, 0, stream>>>(
        f1, f2, Z, nullptr, 1.0f / 16.0f, 0, DIM, DIM, DIM, LDZ,
        (long)NQ * DIM, (long)NQ * DIM, (long)NAUG * LDZ);

    // persistent Sinkhorn (100 iters, device-wide barriers)
    sinkhorn_kernel<<<SINK_BLOCKS, blk, 0, stream>>>(Z, u, v, bar);

    // out = exp(Z + u + v - norm) in place
    const dim3 gf((NAUG + 255) / 256, NAUG, BATCH);
    finalize_kernel<<<gf, blk, 0, stream>>>(Z, u, v);
}
